// QuantizedMultiHeadAttention_17085379903680
// MI455X (gfx1250) — 16-bit and 8-bit matrix operand order repaired and checked by reading, compiled, not run on hardware
//
#include <hip/hip_runtime.h>


typedef __attribute__((ext_vector_type(16))) _Float16 v16h;
typedef __attribute__((ext_vector_type(8)))  _Float16 v8h;
typedef __attribute__((ext_vector_type(8)))  float    v8f;
typedef __attribute__((ext_vector_type(8)))  int      v8i;
typedef __attribute__((ext_vector_type(4)))  int      v4i;
typedef __attribute__((ext_vector_type(2)))  int      v2i;

#define DEV static __device__ __forceinline__

#define NB 2
#define NN 2048
#define NC 768
#define NH 12
#define ND 64
#define BH (NB*NH)
#define M1 (NB*NN)
#define N1 (3*NC)
#define ATT_SCALE 0.125f
#define P_FLOOR 5.9604645e-8f

DEV float wave_max(float v){
#pragma unroll
  for (int m = 16; m > 0; m >>= 1) v = fmaxf(v, __shfl_xor(v, m, 32));
  return v;
}
DEV float half16_max(float v){
#pragma unroll
  for (int m = 8; m > 0; m >>= 1) v = fmaxf(v, __shfl_xor(v, m, 32));
  return v;
}
DEV float half16_sum(float v){
#pragma unroll
  for (int m = 8; m > 0; m >>= 1) v += __shfl_xor(v, m, 32);
  return v;
}
DEV float qclamp(float q){ return fminf(fmaxf(q, -127.f), 127.f); }

DEV v16h cat16(v8h lo, v8h hi){
  return __builtin_shufflevector(lo, hi, 0,1,2,3,4,5,6,7,8,9,10,11,12,13,14,15);
}
DEV v16h load_ah(const _Float16* __restrict__ arow, int k0){
  return cat16(*(const v8h*)(arow + k0), *(const v8h*)(arow + k0 + 16));
}
DEV v16h load_bh(const _Float16* __restrict__ w){
  return cat16(*(const v8h*)w, *(const v8h*)(w + 16));
}
DEV v8i load_a8(const signed char* __restrict__ q, int row0, int lane){
  int r = lane & 15, hf = lane >> 4;
  const signed char* p = q + (size_t)(row0 + r) * ND + hf * 8;
  v2i c0 = *(const v2i*)(p);
  v2i c1 = *(const v2i*)(p + 16);
  v2i c2 = *(const v2i*)(p + 32);
  v2i c3 = *(const v2i*)(p + 48);
  v8i a;
  a[0]=c0[0]; a[1]=c0[1]; a[2]=c1[0]; a[3]=c1[1];
  a[4]=c2[0]; a[5]=c2[1]; a[6]=c3[0]; a[7]=c3[1];
  return a;
}
DEV v8i load_b8(const signed char* __restrict__ k, int col0, int lane){
  int c = lane & 15, hf = lane >> 4;
  const signed char* p = k + (size_t)(col0 + c) * ND + hf * 8;
  v2i c0 = *(const v2i*)(p);
  v2i c1 = *(const v2i*)(p + 16);
  v2i c2 = *(const v2i*)(p + 32);
  v2i c3 = *(const v2i*)(p + 48);
  v8i b;
  b[0]=c0[0]; b[1]=c0[1]; b[2]=c1[0]; b[3]=c1[1];
  b[4]=c2[0]; b[5]=c2[1]; b[6]=c3[0]; b[7]=c3[1];
  return b;
}

__global__ void __launch_bounds__(32) k_init(float* s){
  if (threadIdx.x < 8) s[threadIdx.x] = 0.f;
}

__global__ void __launch_bounds__(256) k_absmax(const float* __restrict__ x, int n,
                                                unsigned* __restrict__ slot){
  float m = 0.f;
  for (int i = blockIdx.x * blockDim.x + threadIdx.x; i < n; i += gridDim.x * blockDim.x)
    m = fmaxf(m, fabsf(x[i]));
  m = wave_max(m);
  if ((threadIdx.x & 31) == 0) atomicMax(slot, __float_as_uint(m));
}

__global__ void __launch_bounds__(256) k_f32_to_f16(const float* __restrict__ x,
                                                    _Float16* __restrict__ y, int n){
  int i = blockIdx.x * blockDim.x + threadIdx.x;
  if (i < n) y[i] = (_Float16)x[i];
}

__global__ void __launch_bounds__(256) k_quant_w(const float* __restrict__ w,
                                                 const float* __restrict__ amax,
                                                 _Float16* __restrict__ wh, int n){
  int i = blockIdx.x * blockDim.x + threadIdx.x;
  if (i >= n) return;
  float s = amax[0] * (1.f/127.f) + 1e-8f;
  wh[i] = (_Float16)(qclamp(rintf(w[i] / s)) * s);
}

__global__ void __launch_bounds__(256) k_final(const float* __restrict__ y,
                                               const float* __restrict__ amax,
                                               float* __restrict__ out, int n){
  int i = blockIdx.x * blockDim.x + threadIdx.x;
  if (i >= n) return;
  float s = amax[0] * (1.f/127.f) + 1e-8f;
  out[i] = qclamp(rintf(y[i] / s)) * s;
}

__global__ void __launch_bounds__(256) k_quant_qkv(const float* __restrict__ qkv,
                                                   const float* __restrict__ scal,
                                                   signed char* __restrict__ q8,
                                                   signed char* __restrict__ k8,
                                                   _Float16* __restrict__ vT){
  int i = blockIdx.x * blockDim.x + threadIdx.x;
  if (i >= BH * NN * ND) return;
  int d  = i & 63;
  int n  = (i >> 6) & (NN - 1);
  int bh = i >> 17;
  int b = bh / NH, h = bh % NH;
  float s = scal[2] * (1.f/127.f) + 1e-8f;
  float invs = 1.f / s;
  size_t base = ((size_t)(b * NN + n)) * N1 + h * ND + d;
  q8[(size_t)(bh * NN + n) * ND + d] = (signed char)(int)qclamp(rintf(qkv[base]        * invs));
  k8[(size_t)(bh * NN + n) * ND + d] = (signed char)(int)qclamp(rintf(qkv[base +   NC] * invs));
  float vi = qclamp(rintf(qkv[base + 2*NC] * invs));
  vT[((size_t)bh * ND + d) * NN + n] = (_Float16)(vi * s);
}

__global__ void __launch_bounds__(128) k_gemm_f16(const _Float16* __restrict__ A,
                                                  const _Float16* __restrict__ W,
                                                  const float* __restrict__ bias,
                                                  float* __restrict__ C,
                                                  int M, int N, int K,
                                                  unsigned* __restrict__ amax){
  int wid  = (blockIdx.x * blockDim.x + threadIdx.x) >> 5;
  int lane = threadIdx.x & 31;
  int ntile = N >> 6;
  int tm = wid / ntile, tn = wid % ntile;
  if (tm * 16 >= M) return;
  int r = lane & 15, hf = lane >> 4;
  const _Float16* arow  = A + (size_t)(tm * 16 + r) * K + hf * 8;
  const _Float16* wrow0 = W + (size_t)(tn * 64 + r) * K + hf * 8;
  v8f acc[4] = {{},{},{},{}};
  v16h a0 = load_ah(arow, 0);
  v16h b0[4];
#pragma unroll
  for (int j = 0; j < 4; ++j) b0[j] = load_bh(wrow0 + (size_t)j * 16 * K);
  v16h a1 = load_ah(arow, 32);
  v16h b1[4];
#pragma unroll
  for (int j = 0; j < 4; ++j) b1[j] = load_bh(wrow0 + (size_t)j * 16 * K + 32);
  for (int k0 = 64; k0 + 64 <= K; k0 += 64) {
#pragma unroll
    for (int j = 0; j < 4; ++j)
      acc[j] = __builtin_amdgcn_wmma_f32_16x16x32_f16(false, a0, false, b0[j],
                                                      (short)0, acc[j], false, false);
    a0 = load_ah(arow, k0);
#pragma unroll
    for (int j = 0; j < 4; ++j) b0[j] = load_bh(wrow0 + (size_t)j * 16 * K + k0);
    __builtin_prefetch(arow + k0 + 192, 0, 1);
#pragma unroll
    for (int j = 0; j < 4; ++j)
      acc[j] = __builtin_amdgcn_wmma_f32_16x16x32_f16(false, a1, false, b1[j],
                                                      (short)0, acc[j], false, false);
    a1 = load_ah(arow, k0 + 32);
#pragma unroll
    for (int j = 0; j < 4; ++j) b1[j] = load_bh(wrow0 + (size_t)j * 16 * K + k0 + 32);
  }
#pragma unroll
  for (int j = 0; j < 4; ++j)
    acc[j] = __builtin_amdgcn_wmma_f32_16x16x32_f16(false, a0, false, b0[j],
                                                    (short)0, acc[j], false, false);
#pragma unroll
  for (int j = 0; j < 4; ++j)
    acc[j] = __builtin_amdgcn_wmma_f32_16x16x32_f16(false, a1, false, b1[j],
                                                    (short)0, acc[j], false, false);
  float mx = 0.f;
#pragma unroll
  for (int j = 0; j < 4; ++j) {
    int col = tn * 64 + j * 16 + r;
    float bv = bias[col];
#pragma unroll
    for (int i = 0; i < 8; ++i) {
      int row = tm * 16 + hf * 8 + i;
      float v = acc[j][i] + bv;
      C[(size_t)row * N + col] = v;
      mx = fmaxf(mx, fabsf(v));
    }
  }
  mx = wave_max(mx);
  if (lane == 0) atomicMax(amax, __float_as_uint(mx));
}

__global__ void __launch_bounds__(128) k_attn_amax(const signed char* __restrict__ q8,
                                                   const signed char* __restrict__ k8,
                                                   const float* __restrict__ scal,
                                                   unsigned* __restrict__ amax){
  int wid  = (blockIdx.x * blockDim.x + threadIdx.x) >> 5;
  int lane = threadIdx.x & 31;
  int strip = wid & 15;
  int mt    = (wid >> 4) & 127;
  int bh    = wid >> 11;
  if (bh >= BH) return;
  float sq = scal[2] * (1.f/127.f) + 1e-8f;
  float cf = sq * sq * ATT_SCALE;
  const signed char* q = q8 + (size_t)bh * NN * ND;
  const signed char* k = k8 + (size_t)bh * NN * ND;
  v8i a = load_a8(q, mt * 16, lane);
  v8i z = {};
  float mx = 0.f;
  v8i b = load_b8(k, strip * 128, lane);
  for (int t = 0; t < 8; ++t) {
    v8i d = __builtin_amdgcn_wmma_i32_16x16x64_iu8(true, a, true, b, z, false, false);
    b = load_b8(k, strip * 128 + ((t + 1) & 7) * 16, lane);
#pragma unroll
    for (int i = 0; i < 8; ++i) mx = fmaxf(mx, fabsf((float)d[i] * cf));
  }
  mx = wave_max(mx);
  if (lane == 0) atomicMax(amax, __float_as_uint(mx));
}

__global__ void __launch_bounds__(128) k_softmax_stats(const signed char* __restrict__ q8,
                                                       const signed char* __restrict__ k8,
                                                       const float* __restrict__ scal,
                                                       float* __restrict__ stats){
  int wid  = (blockIdx.x * blockDim.x + threadIdx.x) >> 5;
  int lane = threadIdx.x & 31;
  int mt = wid & 127;
  int bh = wid >> 7;
  if (bh >= BH) return;
  float sq = scal[2] * (1.f/127.f) + 1e-8f;
  float sa = scal[3] * (1.f/127.f) + 1e-8f;
  float cf = sq * sq * ATT_SCALE, inv_sa = 1.f / sa;
  const signed char* q = q8 + (size_t)bh * NN * ND;
  const signed char* k = k8 + (size_t)bh * NN * ND;
  v8i a = load_a8(q, mt * 16, lane);
  v8i z = {};
  float rowM[8], rowS[8];
#pragma unroll
  for (int i = 0; i < 8; ++i) { rowM[i] = -1e30f; rowS[i] = 0.f; }
  v8i b = load_b8(k, 0, lane);
  for (int t = 0; t < 128; ++t) {
    v8i d = __builtin_amdgcn_wmma_i32_16x16x64_iu8(true, a, true, b, z, false, false);
    b = load_b8(k, ((t + 1) & 127) * 16, lane);
#pragma unroll
    for (int i = 0; i < 8; ++i) {
      float aq = qclamp(rintf((float)d[i] * cf * inv_sa)) * sa;
      float tm = half16_max(aq);
      float nm = fmaxf(rowM[i], tm);
      float ts = half16_sum(__expf(aq - nm));
      rowS[i] = rowS[i] * __expf(rowM[i] - nm) + ts;
      rowM[i] = nm;
    }
  }
  int hf = lane >> 4;
  if ((lane & 15) == 0) {
#pragma unroll
    for (int i = 0; i < 8; ++i) {
      size_t s = ((size_t)bh * NN + mt * 16 + hf * 8 + i) * 2;
      stats[s] = rowM[i]; stats[s + 1] = rowS[i];
    }
  }
}

__global__ void __launch_bounds__(128) k_pv(const signed char* __restrict__ q8,
                                            const signed char* __restrict__ k8,
                                            const _Float16* __restrict__ vT,
                                            const float* __restrict__ scal,
                                            const float* __restrict__ stats,
                                            _Float16* __restrict__ oh){
  __shared__ _Float16 pbuf[4][16 * 32];
  int wl   = threadIdx.x >> 5;
  int wid  = (blockIdx.x * blockDim.x + threadIdx.x) >> 5;
  int lane = threadIdx.x & 31;
  int mt = wid & 127;
  int bh = wid >> 7;
  if (bh >= BH) return;
  int b = bh / NH, h = bh % NH;
  float sq = scal[2] * (1.f/127.f) + 1e-8f;
  float sa = scal[3] * (1.f/127.f) + 1e-8f;
  float cf = sq * sq * ATT_SCALE, inv_sa = 1.f / sa;
  const signed char* q  = q8 + (size_t)bh * NN * ND;
  const signed char* k  = k8 + (size_t)bh * NN * ND;
  const _Float16*    vt = vT + (size_t)bh * ND * NN;
  int r = lane & 15, hf = lane >> 4;
  v8i a = load_a8(q, mt * 16, lane);
  v8i z = {};
  float Mr[8], iSr[8];
#pragma unroll
  for (int i = 0; i < 8; ++i) {
    size_t s = ((size_t)bh * NN + mt * 16 + hf * 8 + i) * 2;
    Mr[i] = stats[s]; iSr[i] = 1.f / stats[s + 1];
  }
  v8f o[4] = {{},{},{},{}};
  _Float16* pt = pbuf[wl];
  v8i bb0 = load_b8(k, 0, lane);
  v8i bb1 = load_b8(k, 16, lane);
  for (int m0 = 0; m0 < NN; m0 += 32) {
    v16h vb[4];
#pragma unroll
    for (int j = 0; j < 4; ++j)
      vb[j] = load_bh(vt + (size_t)(j * 16 + r) * NN + m0 + hf * 8);
    {
      v8i d = __builtin_amdgcn_wmma_i32_16x16x64_iu8(true, a, true, bb0, z, false, false);
      bb0 = load_b8(k, (m0 + 32) & (NN - 1), lane);
#pragma unroll
      for (int i = 0; i < 8; ++i) {
        float aq = qclamp(rintf((float)d[i] * cf * inv_sa)) * sa;
        float p  = __expf(aq - Mr[i]) * iSr[i];
        float lg = fminf(fmaxf(rintf(-log2f(fmaxf(p, P_FLOOR))), 0.f), 15.f);
        pt[(hf * 8 + i) * 32 + r] = (_Float16)exp2f(-lg);
      }
    }
    {
      v8i d = __builtin_amdgcn_wmma_i32_16x16x64_iu8(true, a, true, bb1, z, false, false);
      bb1 = load_b8(k, (m0 + 48) & (NN - 1), lane);
#pragma unroll
      for (int i = 0; i < 8; ++i) {
        float aq = qclamp(rintf((float)d[i] * cf * inv_sa)) * sa;
        float p  = __expf(aq - Mr[i]) * iSr[i];
        float lg = fminf(fmaxf(rintf(-log2f(fmaxf(p, P_FLOOR))), 0.f), 15.f);
        pt[(hf * 8 + i) * 32 + 16 + r] = (_Float16)exp2f(-lg);
      }
    }
    const _Float16* pa0 = pt + r * 32 + hf * 8;
    v16h pa = cat16(*(const v8h*)pa0, *(const v8h*)(pa0 + 16));
#pragma unroll
    for (int j = 0; j < 4; ++j)
      o[j] = __builtin_amdgcn_wmma_f32_16x16x32_f16(false, pa, false, vb[j],
                                                    (short)0, o[j], false, false);
  }
#pragma unroll
  for (int j = 0; j < 4; ++j) {
    int d = h * ND + j * 16 + r;
#pragma unroll
    for (int i = 0; i < 8; ++i) {
      int n = mt * 16 + hf * 8 + i;
      oh[(size_t)(b * NN + n) * NC + d] = (_Float16)o[j][i];
    }
  }
}

extern "C" void kernel_launch(void* const* d_in, const int* in_sizes, int n_in,
                              void* d_out, int out_size, void* d_ws, size_t ws_size,
                              hipStream_t stream){
  const float* x     = (const float*)d_in[0];
  const float* w_qkv = (const float*)d_in[1];
  const float* b_qkv = (const float*)d_in[2];
  const float* w_out = (const float*)d_in[3];
  const float* b_out = (const float*)d_in[4];
  float* out = (float*)d_out;

  char* ws = (char*)d_ws;
  float*    scal  = (float*)ws;
  unsigned* uscal = (unsigned*)ws;
  size_t off = 256;
  _Float16* xh   = (_Float16*)(ws + off);  off += (size_t)M1 * NC * 2;
  _Float16* wqh  = (_Float16*)(ws + off);  off += (size_t)N1 * NC * 2;
  _Float16* woh  = (_Float16*)(ws + off);  off += (size_t)NC * NC * 2;
  float*    qkvf = (float*)   (ws + off);  off += (size_t)M1 * N1 * 4;
  signed char* q8 = (signed char*)(ws + off); off += (size_t)BH * NN * ND;
  signed char* k8 = (signed char*)(ws + off); off += (size_t)BH * NN * ND;
  _Float16* vT   = (_Float16*)(ws + off);  off += (size_t)BH * ND * NN * 2;
  float*    stats= (float*)   (ws + off);  off += (size_t)BH * NN * 2 * 4;
  _Float16* oh   = (_Float16*)(ws + off);  off += (size_t)M1 * NC * 2;
  float*    outf = (float*)   (ws + off);  off += (size_t)M1 * NC * 4;

  const int nX   = M1 * NC;
  const int nWq  = N1 * NC;
  const int nWo  = NC * NC;

  k_init<<<1, 32, 0, stream>>>(scal);
  k_absmax<<<512, 256, 0, stream>>>(w_qkv, nWq, uscal + 0);
  k_absmax<<<512, 256, 0, stream>>>(w_out, nWo, uscal + 1);
  k_f32_to_f16<<<(nX + 255) / 256, 256, 0, stream>>>(x, xh, nX);
  k_quant_w<<<(nWq + 255) / 256, 256, 0, stream>>>(w_qkv, scal + 0, wqh, nWq);
  k_quant_w<<<(nWo + 255) / 256, 256, 0, stream>>>(w_out, scal + 1, woh, nWo);

  k_gemm_f16<<<(M1/16) * (N1/64) / 4, 128, 0, stream>>>(xh, wqh, b_qkv, qkvf,
                                                        M1, N1, NC, uscal + 2);
  k_quant_qkv<<<(BH * NN * ND + 255) / 256, 256, 0, stream>>>(qkvf, scal, q8, k8, vT);

  k_attn_amax<<<BH * 128 * 16 / 4, 128, 0, stream>>>(q8, k8, scal, uscal + 3);
  k_softmax_stats<<<BH * 128 / 4, 128, 0, stream>>>(q8, k8, scal, stats);
  k_pv<<<BH * 128 / 4, 128, 0, stream>>>(q8, k8, vT, scal, stats, oh);

  k_gemm_f16<<<(M1/16) * (NC/64) / 4, 128, 0, stream>>>(oh, woh, b_out, outf,
                                                        M1, NC, NC, uscal + 4);
  k_final<<<(nX + 255) / 256, 256, 0, stream>>>(outf, scal + 4, out, nX);
}
